// Model_2095944040817
// MI455X (gfx1250) — matrix operand order repaired and checked by reading, not compiled, not run on hardware
//
#include <hip/hip_runtime.h>
#include <math.h>

// ---------------------------------------------------------------------------
// Informer forward pass for MI455X (gfx1250), wave32 + WMMA bf16.
// ---------------------------------------------------------------------------

typedef __attribute__((ext_vector_type(16))) __bf16 v16bf;
typedef __attribute__((ext_vector_type(8)))  float  v8f;
typedef __bf16 bf16_t;

#define EPI_NONE  0
#define EPI_GELU  1
#define EPI_BNELU 2

// ---------------------------------------------------------------------------
// GEMM: C[M,N] = epi( (beta?C:0) + A[M,K] @ W[K,N] + bias )
// A rows may be circularly time-shifted within sequences of length Lt
// (kernel-size-3 circular conv realized as 3 accumulated passes).
// Block tile 64x128, 256 threads = 8 waves, each wave owns a 32x32 tile
// (2x2 v_wmma_f32_16x16x32_bf16). Double-buffered bf16 LDS staging:
//   As[p][row][k]  row-major   -> A-frag reads are contiguous (ds_load_b128)
//   Bs[p][n][k]    transposed  -> B-frag reads are contiguous (ds_load_b128)
// W staging is a per-thread column strip so lanes coalesce and the N-guard
// is thread-uniform (no per-element exec-mask branches).
// ---------------------------------------------------------------------------
__global__ __launch_bounds__(256)
void gemm_wmma_kernel(const float* __restrict__ A, const float* __restrict__ W,
                      const float* __restrict__ bias, const float* __restrict__ bng,
                      const float* __restrict__ bnb, float* __restrict__ C,
                      int M, int N, int Kd, int Lt, int shift, int beta, int epi)
{
    __shared__ __align__(16) bf16_t As[2][64][40];   // 2 x 64 rows x 32 K (+8 pad)
    __shared__ __align__(16) bf16_t Bs[2][128][40];  // 2 x 128 n  x 32 K (+8 pad)

    const int tid   = threadIdx.x;
    const int lane  = tid & 31;
    const int wid   = tid >> 5;      // 0..7
    const int waveM = wid >> 2;      // 0..1
    const int waveN = wid & 3;       // 0..3
    const int half  = (lane >> 4) & 1;
    const int l16   = lane & 15;
    const int m0    = blockIdx.y * 64;
    const int n0    = blockIdx.x * 128;

    v8f acc[2][2] = {};

    // A loader: each thread stages 8 consecutive K of one row.
    const int arow = tid >> 2;            // 0..63
    const int acol = (tid & 3) * 8;       // 0,8,16,24
    int gmr = m0 + arow;
    if (gmr >= M) gmr = M - 1;
    int src_row;
    {
        int bidx = gmr / Lt;
        int t    = gmr - bidx * Lt;
        int ts   = t + shift;
        if (ts < 0)   ts += Lt;
        if (ts >= Lt) ts -= Lt;
        src_row = bidx * Lt + ts;
    }
    // B loader: each thread stages a 16-deep K strip of one column n.
    const int bn    = tid & 127;          // 0..127 (consecutive lanes -> coalesced)
    const int bkc   = (tid >> 7) * 16;    // 0 or 16
    const bool bval = (n0 + bn) < N;

    auto stageA = [&](int k0, int p) {
        const float* ap = A + (size_t)src_row * Kd + k0 + acol;
        union { bf16_t h[8]; uint4 q; } u;
        #pragma unroll
        for (int i = 0; i < 8; ++i) u.h[i] = (bf16_t)ap[i];
        *(uint4*)&As[p][arow][acol] = u.q;
    };
    auto stageB = [&](int k0, int p) {
        union { bf16_t h[16]; uint4 q[2]; } u;
        if (bval) {
            const float* wp = W + (size_t)(k0 + bkc) * N + (n0 + bn);
            #pragma unroll
            for (int i = 0; i < 16; ++i) u.h[i] = (bf16_t)wp[(size_t)i * N];
        } else {
            #pragma unroll
            for (int i = 0; i < 16; ++i) u.h[i] = (bf16_t)0.0f;
        }
        uint4* dst = (uint4*)&Bs[p][bn][bkc];
        dst[0] = u.q[0];
        dst[1] = u.q[1];
    };

    stageA(0, 0);
    stageB(0, 0);

    int p = 0;
    for (int k0 = 0; k0 < Kd; k0 += 32) {
        __syncthreads();
        if (k0 + 32 < Kd) {               // overlap next-slab staging with WMMA
            stageA(k0 + 32, p ^ 1);
            stageB(k0 + 32, p ^ 1);
        }
        if (k0 + 64 < Kd) {               // prefetch slab after next
            __builtin_prefetch(A + (size_t)src_row * Kd + k0 + 64 + acol, 0, 1);
            if (bval)
                __builtin_prefetch(W + (size_t)(k0 + 64 + bkc) * N + n0 + bn, 0, 1);
        }

        // Fragments (ISA 16-bit A 16x32 / B 32x16 lane layouts); both sides
        // read two contiguous 16B runs -> ds_load_b128.
        v16bf afrag[2], bfrag[2];
        #pragma unroll
        for (int mt = 0; mt < 2; ++mt) {
            int row = waveM * 32 + mt * 16 + l16;
            #pragma unroll
            for (int i = 0; i < 16; ++i) {
                int kk = (i < 8) ? (8 * half + i) : (16 + 8 * half + (i - 8));
                afrag[mt][i] = As[p][row][kk];
            }
        }
        #pragma unroll
        for (int nt = 0; nt < 2; ++nt) {
            int col = waveN * 32 + nt * 16 + l16;
            #pragma unroll
            for (int i = 0; i < 16; ++i)
                bfrag[nt][i] = Bs[p][col][(i < 8) ? (8 * half + i) : (16 + 8 * half + (i - 8))];
        }
        #pragma unroll
        for (int mt = 0; mt < 2; ++mt)
            #pragma unroll
            for (int nt = 0; nt < 2; ++nt)
                acc[mt][nt] = __builtin_amdgcn_wmma_f32_16x16x32_bf16(
                    false, afrag[mt], false, bfrag[nt], (short)0, acc[mt][nt],
                    false, false);
        p ^= 1;
    }

    const float bnscale = rsqrtf(1.0f + 1e-5f);
    #pragma unroll
    for (int mt = 0; mt < 2; ++mt) {
        #pragma unroll
        for (int nt = 0; nt < 2; ++nt) {
            #pragma unroll
            for (int r = 0; r < 8; ++r) {
                int gm = m0 + waveM * 32 + mt * 16 + r + 8 * half;
                int gn = n0 + waveN * 32 + nt * 16 + l16;
                if (gm < M && gn < N) {
                    float v = acc[mt][nt][r];
                    if (beta) v += C[(size_t)gm * N + gn];
                    if (bias) v += bias[gn];
                    if (epi == EPI_GELU) {
                        v = 0.5f * v * (1.0f + erff(v * 0.70710678118654752f));
                    } else if (epi == EPI_BNELU) {
                        v = v * bnscale * bng[gn] + bnb[gn];
                        v = (v > 0.0f) ? v : (__expf(v) - 1.0f);
                    }
                    C[(size_t)gm * N + gn] = v;
                }
            }
        }
    }
}

// ---------------------------------------------------------------------------
// Flash-style attention: one wave per 16-query tile per (batch,head).
// Q/K/V are [B, L, 512] with head h at columns h*64. Streams keys in chunks
// of 32: S=QK^T via 4 WMMA, online softmax with cross-lane shfl reductions,
// P transposed through LDS into A-fragment layout, O += P@V via 4 WMMA.
// Avoids materializing the 128x720x720 score tensor (265 MB > L2).
// ---------------------------------------------------------------------------
__global__ __launch_bounds__(32)
void attn_kernel(const float* __restrict__ Q, const float* __restrict__ K,
                 const float* __restrict__ V, float* __restrict__ O,
                 int Lq, int Lk, int causal)
{
    __shared__ bf16_t Pl[16][34];

    const int lane = threadIdx.x & 31;
    const int half = lane >> 4;
    const int l16  = lane & 15;
    const int q0   = blockIdx.x * 16;
    const int bh   = blockIdx.y;
    const int b    = bh >> 3, h = bh & 7;

    const float* Qp = Q + ((size_t)b * Lq) * 512 + h * 64;
    const float* Kp = K + ((size_t)b * Lk) * 512 + h * 64;
    const float* Vp = V + ((size_t)b * Lk) * 512 + h * 64;

    // Q fragment (1/sqrt(64) folded in), d split into two K=32 chunks.
    v16bf qa[2];
    {
        int qr = q0 + l16; if (qr >= Lq) qr = Lq - 1;
        const float* qrow = Qp + (size_t)qr * 512;
        #pragma unroll
        for (int c = 0; c < 2; ++c)
            #pragma unroll
            for (int i = 0; i < 16; ++i) {
                int d = c * 32 + ((i < 8) ? (8 * half + i)
                                          : (16 + 8 * half + (i - 8)));
                qa[c][i] = (bf16_t)(qrow[d] * 0.125f);
            }
    }

    v8f ov[4] = {};
    float mr[8], lr[8];
    #pragma unroll
    for (int r = 0; r < 8; ++r) { mr[r] = -1e30f; lr[r] = 0.0f; }

    int nch = (Lk + 31) / 32;
    if (causal) { int lim = (q0 + 47) / 32; if (lim < nch) nch = lim; }

    for (int ch = 0; ch < nch; ++ch) {
        const int k0 = ch * 32;
        // ---- S tile: 16 x 32 as two 16x16 C tiles ----
        v8f s[2];
        #pragma unroll
        for (int nt = 0; nt < 2; ++nt) {
            int key = k0 + nt * 16 + l16;
            bool kin = key < Lk;
            const float* krow = Kp + (size_t)(kin ? key : 0) * 512;
            v8f a = {};
            #pragma unroll
            for (int c = 0; c < 2; ++c) {
                v16bf kb;
                #pragma unroll
                for (int i = 0; i < 16; ++i) {
                    int d = c * 32 + ((i < 8) ? (8 * half + i) : (16 + 8 * half + (i - 8)));
                    kb[i] = kin ? (bf16_t)krow[d] : (bf16_t)0.0f;
                }
                a = __builtin_amdgcn_wmma_f32_16x16x32_bf16(
                        false, qa[c], false, kb, (short)0, a, false, false);
            }
            s[nt] = a;
        }
        // ---- online softmax (element r = row r + 8*half of the lane half)
        #pragma unroll
        for (int r = 0; r < 8; ++r) {
            int qrow = q0 + r + 8 * half;
            float vmax = -1e30f;
            #pragma unroll
            for (int nt = 0; nt < 2; ++nt) {
                int key = k0 + nt * 16 + l16;
                float sv = s[nt][r];
                if (key >= Lk || (causal && key > qrow)) sv = -1e30f;
                s[nt][r] = sv;
                vmax = fmaxf(vmax, sv);
            }
            #pragma unroll
            for (int off = 1; off < 16; off <<= 1)
                vmax = fmaxf(vmax, __shfl_xor(vmax, off, 32));
            float mnew = fmaxf(mr[r], vmax);
            float sc   = __expf(mr[r] - mnew);
            mr[r] = mnew;
            float rs = 0.0f;
            #pragma unroll
            for (int nt = 0; nt < 2; ++nt) {
                float pv = __expf(s[nt][r] - mnew);
                s[nt][r] = pv;
                rs += pv;
            }
            #pragma unroll
            for (int off = 1; off < 16; off <<= 1)
                rs += __shfl_xor(rs, off, 32);
            lr[r] = lr[r] * sc + rs;
            ov[0][r] *= sc; ov[1][r] *= sc; ov[2][r] *= sc; ov[3][r] *= sc;
        }
        // ---- transpose P through LDS into A-fragment layout ----
        #pragma unroll
        for (int nt = 0; nt < 2; ++nt)
            #pragma unroll
            for (int r = 0; r < 8; ++r)
                Pl[r + 8 * half][nt * 16 + l16] = (bf16_t)s[nt][r];
        __syncthreads();
        v16bf pa;
        #pragma unroll
        for (int i = 0; i < 16; ++i) {
            int kk = (i < 8) ? (8 * half + i) : (16 + 8 * half + (i - 8));
            pa[i] = Pl[l16][kk];
        }
        // ---- O += P @ V (K dim = 32 keys, N = 4 tiles of 16 d) ----
        #pragma unroll
        for (int d4 = 0; d4 < 4; ++d4) {
            v16bf vb;
            #pragma unroll
            for (int i = 0; i < 16; ++i) {
                int key = k0 + ((i < 8) ? (8 * half + i) : (16 + 8 * half + (i - 8)));
                vb[i] = (key < Lk) ? (bf16_t)Vp[(size_t)key * 512 + d4 * 16 + l16]
                                   : (bf16_t)0.0f;
            }
            ov[d4] = __builtin_amdgcn_wmma_f32_16x16x32_bf16(
                        false, pa, false, vb, (short)0, ov[d4], false, false);
        }
        __syncthreads();
    }

    #pragma unroll
    for (int r = 0; r < 8; ++r) {
        int gq = q0 + r + 8 * half;
        if (gq >= Lq) continue;
        float inv = 1.0f / lr[r];
        float* orow = O + ((size_t)b * Lq + gq) * 512 + h * 64;
        #pragma unroll
        for (int d4 = 0; d4 < 4; ++d4)
            orow[d4 * 16 + l16] = ov[d4][r] * inv;
    }
}

// ---------------------------------------------------------------------------
// out = LayerNorm(x + res) * g + b  (res may be nullptr). One wave per row.
// ---------------------------------------------------------------------------
__global__ __launch_bounds__(128)
void add_ln_kernel(const float* __restrict__ x, const float* __restrict__ res,
                   const float* __restrict__ g, const float* __restrict__ bta,
                   float* __restrict__ out, int rows)
{
    const int lane = threadIdx.x & 31;
    const int wid  = threadIdx.x >> 5;
    const int row  = blockIdx.x * 4 + wid;
    if (row >= rows) return;
    const size_t base = (size_t)row * 512;
    float v[16];
    float s = 0.0f;
    #pragma unroll
    for (int i = 0; i < 16; ++i) {
        int col = lane + i * 32;
        float t = x[base + col];
        if (res) t += res[base + col];
        v[i] = t; s += t;
    }
    #pragma unroll
    for (int off = 1; off < 32; off <<= 1) s += __shfl_xor(s, off, 32);
    const float mu = s * (1.0f / 512.0f);
    float var = 0.0f;
    #pragma unroll
    for (int i = 0; i < 16; ++i) { float d = v[i] - mu; var += d * d; }
    #pragma unroll
    for (int off = 1; off < 32; off <<= 1) var += __shfl_xor(var, off, 32);
    var *= (1.0f / 512.0f);
    const float inv = rsqrtf(var + 1e-5f);
    #pragma unroll
    for (int i = 0; i < 16; ++i) {
        int col = lane + i * 32;
        out[base + col] = (v[i] - mu) * inv * g[col] + bta[col];
    }
}

// ---------------------------------------------------------------------------
// Embedding: circular conv1d (k=3, c_in=7) + sinusoidal PE + mark @ temp_w
// ---------------------------------------------------------------------------
__global__ void embed_kernel(const float* __restrict__ x, const float* __restrict__ mark,
                             const float* __restrict__ tokw, const float* __restrict__ tempw,
                             float* __restrict__ out, int L)
{
    const int row = blockIdx.x;
    const int b = row / L, t = row % L;
    float xv[21];
    #pragma unroll
    for (int j = 0; j < 3; ++j) {
        int ts = t + j - 1; if (ts < 0) ts += L; if (ts >= L) ts -= L;
        const float* xp = x + ((size_t)b * L + ts) * 7;
        #pragma unroll
        for (int c = 0; c < 7; ++c) xv[j * 7 + c] = xp[c];
    }
    float mk[4];
    const float* mp = mark + ((size_t)b * L + t) * 4;
    #pragma unroll
    for (int f = 0; f < 4; ++f) mk[f] = mp[f];
    const float nl = -logf(10000.0f) / 512.0f;
    for (int d = threadIdx.x; d < 512; d += blockDim.x) {
        int i2 = d & ~1;
        float ang = (float)t * __expf((float)i2 * nl);
        float acc = (d & 1) ? __cosf(ang) : __sinf(ang);
        #pragma unroll
        for (int j = 0; j < 3; ++j)
            #pragma unroll
            for (int c = 0; c < 7; ++c)
                acc += xv[j * 7 + c] * tokw[((size_t)j * 7 + c) * 512 + d];
        #pragma unroll
        for (int f = 0; f < 4; ++f) acc += mk[f] * tempw[(size_t)f * 512 + d];
        out[(size_t)row * 512 + d] = acc;
    }
}

// Max-pool over time: window 3, stride 2, pad 1 (with -inf identity).
__global__ void pool_kernel(const float* __restrict__ src, float* __restrict__ dst,
                            int Lin, int Lout)
{
    int i = blockIdx.x * blockDim.x + threadIdx.x;
    int total = 16 * Lout * 512;
    if (i >= total) return;
    int d = i % 512;
    int r = i / 512;
    int t = r % Lout;
    int b = r / Lout;
    int t0 = 2 * t - 1;
    float m = -3.0e38f;
    #pragma unroll
    for (int j = 0; j < 3; ++j) {
        int tt = t0 + j;
        if (tt >= 0 && tt < Lin) m = fmaxf(m, src[((size_t)b * Lin + tt) * 512 + d]);
    }
    dst[i] = m;
}

// Final slice: out[b, 0:336, 0:7] = proj[b, 336:672, 0:7]
__global__ void slice_kernel(const float* __restrict__ src, float* __restrict__ dst)
{
    int i = blockIdx.x * blockDim.x + threadIdx.x;
    const int total = 16 * 336 * 7;
    if (i >= total) return;
    int c = i % 7;
    int r = i / 7;
    int t = r % 336;
    int b = r / 336;
    dst[i] = src[((size_t)b * 672 + 336 + t) * 7 + c];
}

// ---------------------------------------------------------------------------
// Host orchestration. Param leaves flattened in jax tree order (dict keys
// sorted recursively, lists in order) after the 4 input arrays:
//  4..11  dec.cross {k.b,k.w,o.b,o.w,q.b,q.w,v.b,v.w}
//  12..15 dec.ff1{b,w} ff2{b,w}   16..21 dec.ln1{b,g} ln2{b,g} ln3{b,g}
//  22..29 dec.self {k.b,k.w,o.b,o.w,q.b,q.w,v.b,v.w}
//  30,31  dec_norm{b,g}  32 dec_temp_w  33 dec_tok_w
//  34..37 distil{b,bn_b,bn_g,w}
//  38..53 enc_layer0, 54..69 enc_layer1:
//         {k.b,k.w,o.b,o.w,q.b,q.w,v.b,v.w,ff1.b,ff1.w,ff2.b,ff2.w,ln1.b,ln1.g,ln2.b,ln2.g}
//  70,71  enc_norm{b,g}  72 enc_temp_w  73 enc_tok_w  74 proj.b  75 proj.w
// ---------------------------------------------------------------------------
extern "C" void kernel_launch(void* const* d_in, const int* in_sizes, int n_in,
                              void* d_out, int out_size, void* d_ws, size_t ws_size,
                              hipStream_t stream)
{
    (void)in_sizes; (void)n_in; (void)out_size; (void)ws_size;

    const float* x_enc      = (const float*)d_in[0];
    const float* x_mark_enc = (const float*)d_in[1];
    const float* x_dec      = (const float*)d_in[2];
    const float* x_mark_dec = (const float*)d_in[3];
    auto P = [&](int i) -> const float* { return (const float*)d_in[i]; };

    // Workspace carve (floats): 5*N1 + NF + NE  (~214 MB)
    float* ws = (float*)d_ws;
    const size_t N1 = (size_t)16 * 720 * 512;
    const size_t NF = (size_t)16 * 720 * 2048;
    float* X  = ws;
    float* Y  = X  + N1;
    float* Qb = Y  + N1;
    float* Kb = Qb + N1;
    float* Vb = Kb + N1;
    float* Fb = Vb + N1;
    float* Eo = Fb + NF;

    auto gemm = [&](const float* A, const float* W, const float* bias, float* C,
                    int M, int N, int K, int epi = EPI_NONE, int Lt = 1,
                    int shift = 0, int beta = 0,
                    const float* bng = nullptr, const float* bnb = nullptr) {
        dim3 g((N + 127) / 128, (M + 63) / 64);
        gemm_wmma_kernel<<<g, dim3(256), 0, stream>>>(A, W, bias, bng, bnb, C,
                                                      M, N, K, Lt, shift, beta, epi);
    };
    auto ln = [&](const float* x, const float* res, const float* g,
                  const float* b, float* out, int rows) {
        add_ln_kernel<<<dim3((rows + 3) / 4), dim3(128), 0, stream>>>(x, res, g, b, out, rows);
    };
    auto attn = [&](float* O, int Lq, int Lk, int causal) {
        attn_kernel<<<dim3((Lq + 15) / 16, 128), dim3(32), 0, stream>>>(Qb, Kb, Vb, O, Lq, Lk, causal);
    };
    auto enc_layer = [&](int base, int M, int L) {
        gemm(X, P(base + 5), P(base + 4), Qb, M, 512, 512);   // Q
        gemm(X, P(base + 1), P(base + 0), Kb, M, 512, 512);   // K
        gemm(X, P(base + 7), P(base + 6), Vb, M, 512, 512);   // V
        attn(Y, L, L, 0);
        gemm(Y, P(base + 3), P(base + 2), Qb, M, 512, 512);   // O proj
        ln(X, Qb, P(base + 13), P(base + 12), X, M);          // LN1(x + attn)
        gemm(X,  P(base + 9),  P(base + 8),  Fb, M, 2048, 512, EPI_GELU);
        gemm(Fb, P(base + 11), P(base + 10), Y,  M, 512, 2048);
        ln(X, Y, P(base + 15), P(base + 14), X, M);           // LN2(x + ffn)
    };

    // -------------------- Encoder --------------------
    embed_kernel<<<16 * 720, 128, 0, stream>>>(x_enc, x_mark_enc, P(73), P(72), X, 720);
    enc_layer(38, 16 * 720, 720);

    // Distil: circular conv(k=3) as 3 accumulated shifted GEMMs, BN+ELU fused
    const float* dw = P(37);
    gemm(X, dw,                 nullptr, Fb, 11520, 512, 512, EPI_NONE,  720, -1, 0);
    gemm(X, dw + 512 * 512,     nullptr, Fb, 11520, 512, 512, EPI_NONE,  720,  0, 1);
    gemm(X, dw + 2 * 512 * 512, P(34),   Fb, 11520, 512, 512, EPI_BNELU, 720, +1, 1, P(36), P(35));
    pool_kernel<<<(16 * 360 * 512 + 255) / 256, 256, 0, stream>>>(Fb, X, 720, 360);

    enc_layer(54, 16 * 360, 360);
    ln(X, nullptr, P(71), P(70), Eo, 16 * 360);               // enc_norm

    // -------------------- Decoder --------------------
    embed_kernel<<<16 * 672, 128, 0, stream>>>(x_dec, x_mark_dec, P(33), P(32), X, 672);

    // causal self-attention
    gemm(X, P(27), P(26), Qb, 10752, 512, 512);
    gemm(X, P(23), P(22), Kb, 10752, 512, 512);
    gemm(X, P(29), P(28), Vb, 10752, 512, 512);
    attn(Y, 672, 672, 1);
    gemm(Y, P(25), P(24), Qb, 10752, 512, 512);
    ln(X, Qb, P(17), P(16), X, 10752);

    // cross-attention with encoder output (L=360)
    gemm(X,  P(9),  P(8),  Qb, 10752, 512, 512);
    gemm(Eo, P(5),  P(4),  Kb, 5760,  512, 512);
    gemm(Eo, P(11), P(10), Vb, 5760,  512, 512);
    attn(Y, 672, 360, 0);
    gemm(Y, P(7), P(6), Qb, 10752, 512, 512);
    ln(X, Qb, P(19), P(18), X, 10752);

    // FFN + LN3 + dec_norm
    gemm(X,  P(13), P(12), Fb, 10752, 2048, 512, EPI_GELU);
    gemm(Fb, P(15), P(14), Y,  10752, 512, 2048);
    ln(X, Y, P(21), P(20), X, 10752);
    ln(X, nullptr, P(31), P(30), X, 10752);

    // projection 512 -> 7 and slice of last PRED_LEN steps
    gemm(X, P(75), P(74), Y, 10752, 7, 512);
    slice_kernel<<<(16 * 336 * 7 + 255) / 256, 256, 0, stream>>>(Y, (float*)d_out);
}
